// mha_74234214744688
// MI455X (gfx1250) — matrix operand order repaired and checked by reading, not compiled, not run on hardware
//
#include <hip/hip_runtime.h>

// ---------------------------------------------------------------------------
// MHA forward for MI455X (gfx1250): wave32, v_wmma_f32_16x16x32_bf16,
// TDM (tensor_load_to_lds) staging of GEMM B panels into LDS.
// out  = (x @ w_qkv -> q,k,v) -> causal softmax(QK^T/8) @ V -> @ w_proj
// d_out = [ out (2*2048*1024 f32) | w (2*16*2048*2048 f32) ]
// ---------------------------------------------------------------------------

typedef __attribute__((ext_vector_type(16))) __bf16   v16bf;
typedef __attribute__((ext_vector_type(8)))  float    v8f;
typedef __attribute__((ext_vector_type(4)))  unsigned v4u;
typedef __attribute__((ext_vector_type(8)))  int      v8i;
typedef __attribute__((ext_vector_type(4)))  int      v4i;

#ifndef __has_builtin
#define __has_builtin(x) 0
#endif
#if __has_builtin(__builtin_amdgcn_tensor_load_to_lds)
#define USE_TDM 1
#else
#define USE_TDM 0
#endif

union FragAB {
  v16bf v;
  uint4 q[2];      // 32 bytes
  unsigned u[8];
};

__device__ __forceinline__ unsigned short f32_to_bf16(float f) {
  unsigned int u = __float_as_uint(f);
  u += 0x7FFFu + ((u >> 16) & 1u);   // round-to-nearest-even
  return (unsigned short)(u >> 16);
}

__device__ __forceinline__ unsigned pack_bf16x2(float lo, float hi) {
  return (unsigned)f32_to_bf16(lo) | ((unsigned)f32_to_bf16(hi) << 16);
}

__device__ __forceinline__ v8f zero_v8f() {
  v8f z = {0.f, 0.f, 0.f, 0.f, 0.f, 0.f, 0.f, 0.f};
  return z;
}

#if USE_TDM
// Issue one TDM 2D tile load: 64 rows x 256 bf16 elements, row stride 1024
// elements, from gsrc (row-major) into LDS byte offset lds_off.
__device__ __forceinline__ void tdm_load_b_panel(const unsigned short* gsrc,
                                                 unsigned lds_off) {
  const unsigned long long ga = (unsigned long long)(uintptr_t)gsrc;
  v4u g0;
  g0.x = 1u;                                            // count=1, user D#
  g0.y = lds_off;                                       // LDS byte address
  g0.z = (unsigned)ga;                                  // global_addr[31:0]
  g0.w = (unsigned)((ga >> 32) & 0x01FFFFFFu)           // global_addr[56:32]
         | (2u << 30);                                  // type=2 ("image")
  v8i g1;
  g1[0] = 0x00010000;            // wg_mask=0, data_size=1 (2 bytes), no flags
  g1[1] = (int)(256u << 16);     // tensor_dim0[15:0]=256 in bits[31:16]
  g1[2] = (int)(64u << 16);      // tensor_dim0 hi=0 | tensor_dim1[15:0]=64
  g1[3] = (int)(256u << 16);     // tensor_dim1 hi=0 | tile_dim0=256
  g1[4] = 64;                    // tile_dim1=64, tile_dim2=0
  g1[5] = 1024;                  // tensor_dim0_stride lo32 = 1024 elements
  g1[6] = 0;                     // stride0 hi | stride1 lo
  g1[7] = 0;
  v4i gz = {0, 0, 0, 0};
#if __clang_major__ >= 23
  v8i gz8 = {0, 0, 0, 0, 0, 0, 0, 0};
  __builtin_amdgcn_tensor_load_to_lds(g0, g1, gz, gz, gz8, 0);
#else
  __builtin_amdgcn_tensor_load_to_lds(g0, g1, gz, gz, 0);
#endif
}
#endif

// ---------------------------------------------------------------------------
// Kernel 1: cast x to bf16; cast + transpose weights to [N][K] bf16.
// ---------------------------------------------------------------------------
__global__ void cast_tr_kernel(const float* __restrict__ x,
                               const float* __restrict__ wqkv,
                               const float* __restrict__ wproj,
                               unsigned short* __restrict__ xb,
                               unsigned short* __restrict__ wqkvT,
                               unsigned short* __restrict__ wprojT) {
  const size_t tid    = (size_t)blockIdx.x * blockDim.x + threadIdx.x;
  const size_t stride = (size_t)gridDim.x * blockDim.x;
  for (size_t i = tid; i < (size_t)4096 * 1024; i += stride)
    xb[i] = f32_to_bf16(x[i]);
  for (size_t i = tid; i < (size_t)3072 * 1024; i += stride) {
    const size_t n = i >> 10, k = i & 1023;
    wqkvT[i] = f32_to_bf16(wqkv[k * 3072 + n]);
  }
  for (size_t i = tid; i < (size_t)1024 * 1024; i += stride) {
    const size_t n = i >> 10, k = i & 1023;
    wprojT[i] = f32_to_bf16(wproj[k * 1024 + n]);
  }
}

// ---------------------------------------------------------------------------
// Shared GEMM core: block = 8 waves = M 128 x N 64. B panel (64 rows x 256 K,
// 32 KB) staged in LDS via TDM per 256-wide K chunk; A-fragments streamed from
// global. All 4 B fragments are hoisted into distinct registers so the 8
// ds_load_b128 can clause and overlap the 4 WMMAs.
// ---------------------------------------------------------------------------
#define GEMM_CORE(A_PTR, BT_PTR, NT, MT)                                       \
  extern __shared__ unsigned short lds_b[]; /* 64 x 256 bf16 */                \
  const int wave = threadIdx.x >> 5;                                           \
  const int lane = threadIdx.x & 31;                                           \
  const int hi16 = lane >> 4;                                                  \
  const int l16  = lane & 15;                                                  \
  const int nt   = (NT);                                                       \
  const int mt   = (MT);                                                       \
  const int row  = mt * 16 + l16;                                              \
  const unsigned lds_off = (unsigned)(uintptr_t)&lds_b[0];                     \
  v8f acc[4];                                                                  \
  acc[0] = zero_v8f(); acc[1] = zero_v8f();                                    \
  acc[2] = zero_v8f(); acc[3] = zero_v8f();                                    \
  for (int kc = 0; kc < 1024; kc += 256) {                                     \
    __syncthreads(); /* previous chunk fully consumed */                       \
    TDM_OR_COPY(BT_PTR)                                                        \
    __syncthreads(); /* panel visible to all waves */                          \
    for (int kk = 0; kk < 256; kk += 32) {                                     \
      FragAB a;                                                                \
      const unsigned short* ap = (A_PTR) + (size_t)row * 1024 + kc + kk + hi16 * 8; \
      a.q[0] = *(const uint4*)(ap);                                            \
      a.q[1] = *(const uint4*)(ap + 16);                                       \
      FragAB bfr[4];                                                           \
      _Pragma("unroll")                                                        \
      for (int j = 0; j < 4; ++j) {                                            \
        const unsigned short* bp =                                             \
            lds_b + (size_t)(j * 16 + l16) * 256 + kk + hi16 * 8;              \
        bfr[j].q[0] = *(const uint4*)(bp);                                     \
        bfr[j].q[1] = *(const uint4*)(bp + 16);                                \
      }                                                                        \
      _Pragma("unroll")                                                        \
      for (int j = 0; j < 4; ++j)                                              \
        acc[j] = __builtin_amdgcn_wmma_f32_16x16x32_bf16(                      \
            false, a.v, false, bfr[j].v, (short)0, acc[j], false, false);      \
    }                                                                          \
  }

#if USE_TDM
#define TDM_OR_COPY(BT_PTR)                                                    \
    if (wave == 0) {                                                           \
      tdm_load_b_panel((BT_PTR) + (size_t)nt * 64 * 1024 + kc, lds_off);       \
      __builtin_amdgcn_s_wait_tensorcnt(0);                                    \
    }
#else
#define TDM_OR_COPY(BT_PTR)                                                    \
    for (int t = threadIdx.x; t < 2048; t += 256) {                            \
      const int r = t >> 5, c = t & 31; /* 32 uint4 per 256-elem row */        \
      ((uint4*)lds_b)[r * 32 + c] =                                            \
          *(const uint4*)((BT_PTR) + (size_t)(nt * 64 + r) * 1024 + kc + c * 8); \
    }
#endif

// ---------------------------------------------------------------------------
// Kernel 2: QKV GEMM. A = xb [4096][1024], Bt = wqkvT [3072][1024].
// Epilogue scatters into Q[b,h,s,64], K[b,h,s,64], Vt[b,h,64,s] (bf16).
// grid = (32 M-blocks of 128 rows, 48 N-tiles of 64), block = 256.
// ---------------------------------------------------------------------------
__global__ void __launch_bounds__(256)
gemm_qkv_kernel(const unsigned short* __restrict__ A,
                const unsigned short* __restrict__ Bt,
                unsigned short* __restrict__ Qb,
                unsigned short* __restrict__ Kb,
                unsigned short* __restrict__ Vt) {
  GEMM_CORE(A, Bt, blockIdx.y, blockIdx.x * 8 + wave)

#pragma unroll
  for (int j = 0; j < 4; ++j) {
    const int n   = nt * 64 + j * 16 + l16;
    const int t   = n >> 10;          // 0=q 1=k 2=v
    const int rem = n & 1023;
    const int h   = rem >> 6;
    const int d   = rem & 63;
#pragma unroll
    for (int r = 0; r < 8; ++r) {
      const int m = mt * 16 + r + hi16 * 8;
      const int b = m >> 11;
      const int s = m & 2047;
      const unsigned short val = f32_to_bf16(acc[j][r]);
      const size_t bh = (size_t)b * 16 + h;
      if (t == 0)      Qb[(bh * 2048 + s) * 64 + d] = val;
      else if (t == 1) Kb[(bh * 2048 + s) * 64 + d] = val;
      else             Vt[(bh * 64 + d) * 2048 + s] = val;
    }
  }
}

// ---------------------------------------------------------------------------
// Kernel 4: output projection. A = o bf16 [4096][1024], Bt = wprojT
// [1024][1024], C = f32 out [4096][1024]. grid = (32, 16), block = 256.
// ---------------------------------------------------------------------------
__global__ void __launch_bounds__(256)
gemm_proj_kernel(const unsigned short* __restrict__ A,
                 const unsigned short* __restrict__ Bt,
                 float* __restrict__ C) {
  GEMM_CORE(A, Bt, blockIdx.y, blockIdx.x * 8 + wave)

#pragma unroll
  for (int j = 0; j < 4; ++j) {
    const int n = nt * 64 + j * 16 + l16;
#pragma unroll
    for (int r = 0; r < 8; ++r) {
      const int mrow = mt * 16 + r + hi16 * 8;
      C[(size_t)mrow * 1024 + n] = acc[j][r];
    }
  }
}

// ---------------------------------------------------------------------------
// Kernel 3: causal attention, one wave per (b, h, 16-row q-tile).
// Pass 1: online row max/sum over causal K tiles. Pass 2: recompute S,
// stage normalized probs in an LDS f32 panel (row-major float4 stores to w,
// bf16 A-fragment repack for P @ V), zero-fill the upper triangle itself.
// ---------------------------------------------------------------------------
__device__ __forceinline__ v8f qk_tile(const unsigned short* __restrict__ Kh,
                                       const FragAB& aq0, const FragAB& aq1,
                                       int kt, int l16, int hi16, int qbase) {
  FragAB bk0, bk1;
  const unsigned short* kp = Kh + (size_t)(kt * 16 + l16) * 64 + hi16 * 8;
  bk0.q[0] = *(const uint4*)(kp);
  bk0.q[1] = *(const uint4*)(kp + 16);
  bk1.q[0] = *(const uint4*)(kp + 32);
  bk1.q[1] = *(const uint4*)(kp + 48);
  v8f s = zero_v8f();
  s = __builtin_amdgcn_wmma_f32_16x16x32_bf16(false, aq0.v, false, bk0.v,
                                              (short)0, s, false, false);
  s = __builtin_amdgcn_wmma_f32_16x16x32_bf16(false, aq1.v, false, bk1.v,
                                              (short)0, s, false, false);
  const int kpos = kt * 16 + l16;
#pragma unroll
  for (int r = 0; r < 8; ++r) {
    const int qpos = qbase + r + hi16 * 8;
    const float v = s[r] * 0.125f;                 // 1/sqrt(64)
    s[r] = (kpos <= qpos) ? v : -__builtin_inff();
  }
  return s;
}

__global__ void attn_kernel(const unsigned short* __restrict__ Q,
                            const unsigned short* __restrict__ K,
                            const unsigned short* __restrict__ Vt,
                            unsigned short* __restrict__ O,
                            float* __restrict__ Wout) {
  const int lane = threadIdx.x & 31;
  const int hi16 = lane >> 4;
  const int l16  = lane & 15;
  const int qt = blockIdx.x;   // 0..127
  const int h  = blockIdx.y;   // 0..15
  const int b  = blockIdx.z;   // 0..1
  const size_t bh = (size_t)b * 16 + h;
  const unsigned short* Qh = Q  + bh * 2048 * 64;
  const unsigned short* Kh = K  + bh * 2048 * 64;
  const unsigned short* Vh = Vt + bh * 64 * 2048;
  float* Wh = Wout + bh * 2048 * 2048;
  const int qbase = qt * 16;

  FragAB aq0, aq1;   // Q fragments d=0..31 / d=32..63, live in registers
  {
    const unsigned short* qp = Qh + (size_t)(qbase + l16) * 64 + hi16 * 8;
    aq0.q[0] = *(const uint4*)(qp);
    aq0.q[1] = *(const uint4*)(qp + 16);
    aq1.q[0] = *(const uint4*)(qp + 32);
    aq1.q[1] = *(const uint4*)(qp + 48);
  }

  // -------- Zero-fill this q-tile's strictly-upper-triangle columns --------
  const int nkt = qt + 1;
  {
    float4 z4; z4.x = 0.f; z4.y = 0.f; z4.z = 0.f; z4.w = 0.f;
    const int kz = nkt * 16;                      // 64B aligned
#pragma unroll 1
    for (int rr = 0; rr < 16; ++rr) {
      float* wrow = Wh + (size_t)(qbase + rr) * 2048;
      for (int c = kz + lane * 4; c < 2048; c += 128)
        *(float4*)&wrow[c] = z4;
    }
  }

  // -------- Pass 1: softmax statistics --------
  float m[8], l[8];
#pragma unroll
  for (int r = 0; r < 8; ++r) { m[r] = -__builtin_inff(); l[r] = 0.f; }

  for (int kt = 0; kt <= qt; ++kt) {
    v8f s = qk_tile(Kh, aq0, aq1, kt, l16, hi16, qbase);
    float tmax[8];
#pragma unroll
    for (int r = 0; r < 8; ++r) tmax[r] = s[r];
    for (int off = 1; off < 16; off <<= 1) {
#pragma unroll
      for (int r = 0; r < 8; ++r)
        tmax[r] = fmaxf(tmax[r], __shfl_xor(tmax[r], off, 32));
    }
    float mn[8], e[8];
#pragma unroll
    for (int r = 0; r < 8; ++r) {
      mn[r] = fmaxf(m[r], tmax[r]);
      e[r]  = __expf(s[r] - mn[r]);     // exp(-inf) = 0 for masked cols
    }
    for (int off = 1; off < 16; off <<= 1) {
#pragma unroll
      for (int r = 0; r < 8; ++r) e[r] += __shfl_xor(e[r], off, 32);
    }
#pragma unroll
    for (int r = 0; r < 8; ++r) {
      l[r] = l[r] * __expf(m[r] - mn[r]) + e[r];
      m[r] = mn[r];
    }
  }

  float invl[8];
#pragma unroll
  for (int r = 0; r < 8; ++r) invl[r] = 1.0f / l[r];

  // -------- Pass 2: probs out + O = P @ V --------
  __shared__ __align__(16) float pf[16][32];  // 16 q-rows x 32 k-cols (f32)
  v8f oacc[4];
  oacc[0] = zero_v8f(); oacc[1] = zero_v8f();
  oacc[2] = zero_v8f(); oacc[3] = zero_v8f();

  const int npair = (nkt + 1) >> 1;
  for (int ktp = 0; ktp < npair; ++ktp) {
#pragma unroll
    for (int sub = 0; sub < 2; ++sub) {
      const int kt = ktp * 2 + sub;
      if (kt < nkt) {
        v8f s = qk_tile(Kh, aq0, aq1, kt, l16, hi16, qbase);
#pragma unroll
        for (int r = 0; r < 8; ++r)
          pf[r + hi16 * 8][sub * 16 + l16] = __expf(s[r] - m[r]) * invl[r];
      } else {
#pragma unroll
        for (int r = 0; r < 8; ++r)
          pf[r + hi16 * 8][sub * 16 + l16] = 0.f;
      }
    }
    __syncthreads();

    // Coalesced row-major float4 stores of the prob panel to w.
    {
      const int rr = lane >> 3, c4 = (lane & 7) * 4;
#pragma unroll
      for (int g = 0; g < 4; ++g) {
        const int row4 = rr + g * 4;
        *(float4*)&Wh[(size_t)(qbase + row4) * 2048 + ktp * 32 + c4] =
            *(const float4*)&pf[row4][c4];
      }
    }

    // Repack P panel into a bf16 16x32 A-fragment.
    FragAB pa;
    {
      const float4 f0 = *(const float4*)&pf[l16][hi16 * 8 + 0];
      const float4 f1 = *(const float4*)&pf[l16][hi16 * 8 + 4];
      const float4 f2 = *(const float4*)&pf[l16][hi16 * 8 + 16];
      const float4 f3 = *(const float4*)&pf[l16][hi16 * 8 + 20];
      pa.u[0] = pack_bf16x2(f0.x, f0.y);
      pa.u[1] = pack_bf16x2(f0.z, f0.w);
      pa.u[2] = pack_bf16x2(f1.x, f1.y);
      pa.u[3] = pack_bf16x2(f1.z, f1.w);
      pa.u[4] = pack_bf16x2(f2.x, f2.y);
      pa.u[5] = pack_bf16x2(f2.z, f2.w);
      pa.u[6] = pack_bf16x2(f3.x, f3.y);
      pa.u[7] = pack_bf16x2(f3.z, f3.w);
    }

    FragAB bv[4];
#pragma unroll
    for (int dj = 0; dj < 4; ++dj) {
      const unsigned short* vp =
          Vh + (size_t)(dj * 16 + l16) * 2048 + ktp * 32 + hi16 * 8;
      bv[dj].q[0] = *(const uint4*)(vp);
      bv[dj].q[1] = *(const uint4*)(vp + 16);
    }
#pragma unroll
    for (int dj = 0; dj < 4; ++dj)
      oacc[dj] = __builtin_amdgcn_wmma_f32_16x16x32_bf16(
          false, pa.v, false, bv[dj].v, (short)0, oacc[dj], false, false);
    __syncthreads();
  }

#pragma unroll
  for (int dj = 0; dj < 4; ++dj) {
#pragma unroll
    for (int r = 0; r < 8; ++r) {
      const int qpos = qbase + r + hi16 * 8;
      O[(size_t)(b * 2048 + qpos) * 1024 + h * 64 + dj * 16 + l16] =
          f32_to_bf16(oacc[dj][r]);
    }
  }
}

// ---------------------------------------------------------------------------
// Launch
// ---------------------------------------------------------------------------
extern "C" void kernel_launch(void* const* d_in, const int* in_sizes, int n_in,
                              void* d_out, int out_size, void* d_ws, size_t ws_size,
                              hipStream_t stream) {
  const float* x      = (const float*)d_in[0];
  const float* w_qkv  = (const float*)d_in[1];
  const float* w_proj = (const float*)d_in[2];

  float* out   = (float*)d_out;                       // 4096*1024
  float* wattn = out + (size_t)4096 * 1024;           // 2*16*2048*2048

  unsigned short* ws     = (unsigned short*)d_ws;
  unsigned short* xb     = ws;                                   // 4096*1024
  unsigned short* wqkvT  = xb    + (size_t)4096 * 1024;          // 3072*1024
  unsigned short* wprojT = wqkvT + (size_t)3072 * 1024;          // 1024*1024
  unsigned short* Qb     = wprojT + (size_t)1024 * 1024;         // 2*16*2048*64
  unsigned short* Kb     = Qb + (size_t)2 * 16 * 2048 * 64;
  unsigned short* Vt     = Kb + (size_t)2 * 16 * 2048 * 64;
  unsigned short* Ob     = Vt + (size_t)2 * 16 * 2048 * 64;      // 4096*1024

  cast_tr_kernel<<<4096, 256, 0, stream>>>(x, w_qkv, w_proj, xb, wqkvT, wprojT);

  // QKV GEMM: 32 M-blocks x 48 N-tiles, 8 waves/block, 32 KB LDS B panel.
  gemm_qkv_kernel<<<dim3(32, 48), 256, 32768, stream>>>(xb, wqkvT, Qb, Kb, Vt);

  // Attention writes the entire w region itself (lower tiles + zero fill).
  attn_kernel<<<dim3(128, 16, 2), 32, 0, stream>>>(Qb, Kb, Vt, Ob, wattn);

  // Projection GEMM: 32 M-blocks x 16 N-tiles.
  gemm_proj_kernel<<<dim3(32, 16), 256, 32768, stream>>>(Ob, wprojT, out);
}
